// QBNNTransformerBlock_2980707304152
// MI455X (gfx1250) — compile-verified, matrix operand order checked by reading, not run on hardware
//
#include <hip/hip_runtime.h>
#include <math.h>

// ---------------------------------------------------------------------------
// Types for CDNA5 WMMA (wave32, 16x16x32 bf16 -> f32)
// ---------------------------------------------------------------------------
typedef __attribute__((ext_vector_type(16))) __bf16 v16bf;
typedef __attribute__((ext_vector_type(8)))  __bf16 v8bf;
typedef __attribute__((ext_vector_type(4)))  __bf16 v4bf;
typedef __attribute__((ext_vector_type(8)))  float  v8f;
typedef __attribute__((ext_vector_type(4)))  unsigned tdm_g0_t;
typedef __attribute__((ext_vector_type(8)))  unsigned tdm_g1_t;

__device__ inline v8f vzero8() {
  v8f z = {0.f,0.f,0.f,0.f,0.f,0.f,0.f,0.f};
  return z;
}

__device__ inline v8f wmma_bf16(v16bf a, v16bf b, v8f c) {
  // (neg_a, A, neg_b, B, c_mod, C, reuse_a, reuse_b)
  return __builtin_amdgcn_wmma_f32_16x16x32_bf16(false, a, false, b, (short)0, c, false, false);
}

// Load a 16x32 bf16 fragment (A layout; B uses same layout on an N-major tile).
__device__ inline v16bf frag_ld(const __bf16* p, int ld) {
  int lane = threadIdx.x & 31;
  const __bf16* q = p + (lane & 15) * ld + ((lane >> 4) << 3);
  union { v16bf v; v8bf h[2]; } u;
  u.h[0] = *(const v8bf*)(q);
  u.h[1] = *(const v8bf*)(q + 16);
  return u.v;
}

__device__ inline float gelu_f(float x) {
  return 0.5f * x * (1.f + erff(x * 0.70710678118654752f));
}

// LDS byte offset of a generic pointer to __shared__ (flat LDS addr low 32 bits)
__device__ inline unsigned lds_off(const void* p) {
  return (unsigned)(unsigned long long)p;
}

// ---------------------------------------------------------------------------
// TDM: issue a 2D tensor_load_to_lds of tile (tile1 rows x tile0 f32 elems),
// row stride = rowStride elements, OOB-guarded by (tdim0, tdim1).
// Must be issued by exactly one wave; completion via s_wait_tensorcnt.
// ---------------------------------------------------------------------------
__device__ inline void tdm_load_2d_f32(const float* gptr, unsigned ldsAddr,
                                       unsigned tile0, unsigned tile1,
                                       unsigned tdim0, unsigned tdim1,
                                       unsigned long long rowStride) {
  unsigned long long ga = (unsigned long long)gptr;
  unsigned a0 = 1u;                                            // count=1, user desc
  unsigned a1 = ldsAddr;                                       // lds_addr (bytes)
  unsigned a2 = (unsigned)(ga & 0xffffffffull);                // global_addr[31:0]
  unsigned a3 = (unsigned)((ga >> 32) & 0x1ffffffull) | (2u << 30); // addr[56:32] | type=2
  unsigned b0 = (2u << 16);                                    // data_size = 4 bytes
  unsigned b1 = (tdim0 & 0xffffu) << 16;                       // tensor_dim0[15:0]
  unsigned b2 = ((tdim0 >> 16) & 0xffffu) | ((tdim1 & 0xffffu) << 16);
  unsigned b3 = ((tdim1 >> 16) & 0xffffu) | ((tile0 & 0xffffu) << 16); // tile_dim0
  unsigned b4 = (tile1 & 0xffffu);                             // tile_dim1 (tile_dim2=0)
  unsigned b5 = (unsigned)(rowStride & 0xffffffffull);         // tensor_dim0_stride[31:0]
  unsigned b6 = (unsigned)((rowStride >> 32) & 0xffffull);     // stride[47:32]
  unsigned b7 = 0u;

  a0 = (unsigned)__builtin_amdgcn_readfirstlane((int)a0);
  a1 = (unsigned)__builtin_amdgcn_readfirstlane((int)a1);
  a2 = (unsigned)__builtin_amdgcn_readfirstlane((int)a2);
  a3 = (unsigned)__builtin_amdgcn_readfirstlane((int)a3);
  b0 = (unsigned)__builtin_amdgcn_readfirstlane((int)b0);
  b1 = (unsigned)__builtin_amdgcn_readfirstlane((int)b1);
  b2 = (unsigned)__builtin_amdgcn_readfirstlane((int)b2);
  b3 = (unsigned)__builtin_amdgcn_readfirstlane((int)b3);
  b4 = (unsigned)__builtin_amdgcn_readfirstlane((int)b4);
  b5 = (unsigned)__builtin_amdgcn_readfirstlane((int)b5);
  b6 = (unsigned)__builtin_amdgcn_readfirstlane((int)b6);
  b7 = (unsigned)__builtin_amdgcn_readfirstlane((int)b7);

  tdm_g0_t g0; g0[0] = a0; g0[1] = a1; g0[2] = a2; g0[3] = a3;
  tdm_g1_t g1; g1[0] = b0; g1[1] = b1; g1[2] = b2; g1[3] = b3;
               g1[4] = b4; g1[5] = b5; g1[6] = b6; g1[7] = b7;

  asm volatile("tensor_load_to_lds %0, %1" :: "s"(g0), "s"(g1) : "memory");
}

// ---------------------------------------------------------------------------
// Block reduction (sum, sumsq) for 256-thread blocks
// ---------------------------------------------------------------------------
__device__ inline void block_reduce2(float& s, float& ss) {
  #pragma unroll
  for (int o = 16; o > 0; o >>= 1) {
    s  += __shfl_xor(s,  o);
    ss += __shfl_xor(ss, o);
  }
  __shared__ float as_[8], bs_[8];
  int wave = threadIdx.x >> 5, lane = threadIdx.x & 31;
  if (lane == 0) { as_[wave] = s; bs_[wave] = ss; }
  __syncthreads();
  if (threadIdx.x == 0) {
    float ts = 0.f, tss = 0.f;
    for (int i = 0; i < (int)(blockDim.x >> 5); i++) { ts += as_[i]; tss += bs_[i]; }
    as_[0] = ts; bs_[0] = tss;
  }
  __syncthreads();
  s = as_[0]; ss = bs_[0];
}

// ---------------------------------------------------------------------------
// LayerNorm over last dim N (one block per row)
// ---------------------------------------------------------------------------
__global__ __launch_bounds__(256)
void ln_kernel(const float* __restrict__ x, const float* __restrict__ w,
               const float* __restrict__ b, float* __restrict__ y, int N) {
  long long row = blockIdx.x;
  const float* xr = x + row * N;
  float* yr = y + row * N;
  float s = 0.f, ss = 0.f;
  for (int c = threadIdx.x; c < N; c += 256) {
    float v = xr[c]; s += v; ss += v * v;
  }
  block_reduce2(s, ss);
  float mean = s / (float)N;
  float var  = ss / (float)N - mean * mean;
  float inv  = rsqrtf(var + 1e-5f);
  for (int c = threadIdx.x; c < N; c += 256)
    yr[c] = (xr[c] - mean) * inv * w[c] + b[c];
}

// ---------------------------------------------------------------------------
// Generic tiled WMMA GEMM:  C[MxN] = epi(op(A)[MxK] @ W[KxN] + bias (+resid))
// A tile staged via Tensor Data Mover (global -> LDS f32), then converted to
// bf16 in LDS. B tile staged transposed (N-major) with on-the-fly conversion.
// AOP: 0 = identity, 1 = tanh on A elements.
// EPI: 0 = +bias, 1 = gelu(+bias), 2 = resid + (+bias)
// ---------------------------------------------------------------------------
#define TM 128
#define TN 128
#define TK 32

template<int AOP, int EPI>
__global__ __launch_bounds__(256)
void gemm_wmma_kernel(const float* __restrict__ A, const float* __restrict__ W,
                      const float* __restrict__ bias, const float* __restrict__ resid,
                      float* __restrict__ C,
                      int Mr, int Nr, int Kr, int lda, int ldw, int ldc,
                      int batDiv, long long aStr1, long long aStr2,
                      long long wStr, long long cStr) {
  const int z = blockIdx.z;
  A += (long long)(z / batDiv) * aStr1 + (long long)(z % batDiv) * aStr2;
  W += (long long)(z % batDiv) * wStr;
  C += (long long)z * cStr;
  if (resid) resid += (long long)z * cStr;

  __shared__ __align__(16) float  Af32[TM * TK];  // 16 KB TDM staging (f32)
  __shared__ __align__(16) __bf16 As[TM * TK];    // 8 KB row-major MxK bf16
  __shared__ __align__(16) __bf16 Bs[TN * TK];    // 8 KB N-major NxK bf16

  const int tid  = threadIdx.x;
  const int lane = tid & 31;
  const int wave = tid >> 5;
  const int wm   = wave >> 2;   // 0..1  (64 rows each)
  const int wn   = wave & 3;    // 0..3  (32 cols each)
  const int m0   = blockIdx.y * TM;
  const int n0   = blockIdx.x * TN;
  const unsigned afOff = lds_off((const void*)Af32);

  v8f acc[4][2];
  #pragma unroll
  for (int i = 0; i < 4; i++)
    #pragma unroll
    for (int j = 0; j < 2; j++) acc[i][j] = vzero8();

  for (int k0 = 0; k0 < Kr; k0 += TK) {
    // --- stage B tile transposed (N-major 128x32), 16 scalars per thread ---
    // prefetch next K-slab of W while we are at it
    if (k0 + TK < Kr)
      __builtin_prefetch((const void*)(W + (long long)(k0 + TK) * ldw + n0 + (tid & 127)), 0, 0);
    #pragma unroll
    for (int i = 0; i < 16; i++) {
      int e  = tid + i * 256;
      int kk = e >> 7;
      int n  = e & 127;
      int gk = k0 + kk, gn = n0 + n;
      float v = 0.f;
      if (gk < Kr && gn < Nr) v = W[(long long)gk * ldw + gn];
      Bs[n * TK + kk] = (__bf16)v;
    }
    // --- TDM: DMA the A f32 tile (TM x TK, row stride lda) into LDS ---
    if (wave == 0) {
      tdm_load_2d_f32(A + (long long)m0 * lda + k0, afOff,
                      TK, TM,
                      (unsigned)(Kr - k0), (unsigned)(Mr - m0),
                      (unsigned long long)lda);
      __builtin_amdgcn_s_wait_tensorcnt(0);
    }
    __syncthreads();
    // --- convert staged A tile f32 -> bf16 (optional tanh) ---
    #pragma unroll
    for (int i = 0; i < 4; i++) {
      int e  = tid + i * 256;
      int r  = e >> 3;
      int c4 = e & 7;
      float4 v = *(const float4*)&Af32[r * TK + c4 * 4];
      if (AOP == 1) { v.x = tanhf(v.x); v.y = tanhf(v.y); v.z = tanhf(v.z); v.w = tanhf(v.w); }
      v4bf p;
      p[0] = (__bf16)v.x; p[1] = (__bf16)v.y; p[2] = (__bf16)v.z; p[3] = (__bf16)v.w;
      *(v4bf*)&As[r * TK + c4 * 4] = p;
    }
    __syncthreads();

    // --- WMMA compute: each wave 64x32 = 4x2 fragments ---
    v16bf bfr[2];
    #pragma unroll
    for (int nf = 0; nf < 2; nf++)
      bfr[nf] = frag_ld(&Bs[(wn * 32 + nf * 16) * TK], TK);
    #pragma unroll
    for (int mf = 0; mf < 4; mf++) {
      v16bf afr = frag_ld(&As[(wm * 64 + mf * 16) * TK], TK);
      acc[mf][0] = wmma_bf16(afr, bfr[0], acc[mf][0]);
      acc[mf][1] = wmma_bf16(afr, bfr[1], acc[mf][1]);
    }
    __syncthreads();
  }

  // --- epilogue ---
  #pragma unroll
  for (int mf = 0; mf < 4; mf++) {
    #pragma unroll
    for (int nf = 0; nf < 2; nf++) {
      int nb = n0 + wn * 32 + nf * 16 + (lane & 15);
      int mb = m0 + wm * 64 + mf * 16 + ((lane >> 4) << 3);
      if (nb < Nr) {
        float bv = bias ? bias[nb] : 0.f;
        #pragma unroll
        for (int i = 0; i < 8; i++) {
          int m = mb + i;
          if (m < Mr) {
            float v = acc[mf][nf][i] + bv;
            if (EPI == 1) v = gelu_f(v);
            if (EPI == 2) v += resid[(long long)m * ldc + nb];
            C[(long long)m * ldc + nb] = v;
          }
        }
      }
    }
  }
}

// ---------------------------------------------------------------------------
// Flash-style causal attention with QBNN delta term.
// scores = (Q K^T)/8 + lam * (QJ tanh(K)^T); softmax; @V
// Block: 128 threads (4 waves); each block = one 64-row query tile of one (b,h).
// ---------------------------------------------------------------------------
__global__ __launch_bounds__(128)
void attn_kernel(const float* __restrict__ Qg, const float* __restrict__ QJg,
                 const float* __restrict__ Kg, const float* __restrict__ Vg,
                 const float* __restrict__ lam_p, float* __restrict__ ctx) {
  const int S = 1024, E = 1024, Dh = 64, H = 16;
  const int z  = blockIdx.y;
  const int b  = z / H;
  const int hh = z % H;
  const int q0 = blockIdx.x * 64;
  const int tid  = threadIdx.x;
  const int lane = tid & 31;
  const int wv   = tid >> 5;           // wave 0..3 -> query rows [wv*16, wv*16+16)
  const float lam = *lam_p;
  const float scl = 0.125f;            // 1/sqrt(64)

  __shared__ __align__(16) unsigned char smem[58112];
  __bf16* Qs  = (__bf16*)(smem);            // 64x64 q-major  (8 KB)
  __bf16* QJs = (__bf16*)(smem + 8192);     // 64x64 q-major  (8 KB)
  __bf16* Ks  = (__bf16*)(smem + 16384);    // 64x64 kv-major (8 KB)  raw K
  __bf16* Kts = (__bf16*)(smem + 24576);    // 64x64 kv-major (8 KB)  tanh(K)
  __bf16* Vts = (__bf16*)(smem + 32768);    // 64x64 d-major  (8 KB)  V^T
  float*  Ss  = (float* )(smem + 40960);    // 64x64 scores f32 (16 KB)
  float*  m_s = (float* )(smem + 57344);
  float*  l_s = m_s + 64;
  float*  sc_s = m_s + 128;
  __bf16* Ps  = Ks;                          // alias: probs reuse Ks after scores

  const float* Qbase  = Qg  + (long long)(b * S) * E + hh * Dh;
  const float* QJbase = QJg + (long long)z * S * Dh;
  const float* Kbase  = Kg  + (long long)(b * S) * E + hh * Dh;
  const float* Vbase  = Vg  + (long long)(b * S) * E + hh * Dh;

  #pragma unroll
  for (int i = 0; i < 8; i++) {
    int e  = tid + i * 128;
    int r  = e >> 4, c4 = e & 15;
    float4 v = *(const float4*)(Qbase + (long long)(q0 + r) * E + c4 * 4);
    v4bf pq; pq[0] = (__bf16)v.x; pq[1] = (__bf16)v.y; pq[2] = (__bf16)v.z; pq[3] = (__bf16)v.w;
    *(v4bf*)&Qs[r * 64 + c4 * 4] = pq;
    float4 w = *(const float4*)(QJbase + (long long)(q0 + r) * Dh + c4 * 4);
    v4bf pw; pw[0] = (__bf16)w.x; pw[1] = (__bf16)w.y; pw[2] = (__bf16)w.z; pw[3] = (__bf16)w.w;
    *(v4bf*)&QJs[r * 64 + c4 * 4] = pw;
  }
  if (tid < 64) { m_s[tid] = -3.0e38f; l_s[tid] = 0.f; }

  v8f acc[4];
  #pragma unroll
  for (int nf = 0; nf < 4; nf++) acc[nf] = vzero8();
  __syncthreads();

  for (int j0 = 0; j0 <= q0; j0 += 64) {
    #pragma unroll
    for (int i = 0; i < 8; i++) {
      int e  = tid + i * 128;
      int r  = e >> 4, c4 = e & 15;
      float4 v = *(const float4*)(Kbase + (long long)(j0 + r) * E + c4 * 4);
      v4bf pr, pt;
      pr[0] = (__bf16)v.x; pr[1] = (__bf16)v.y; pr[2] = (__bf16)v.z; pr[3] = (__bf16)v.w;
      pt[0] = (__bf16)tanhf(v.x); pt[1] = (__bf16)tanhf(v.y);
      pt[2] = (__bf16)tanhf(v.z); pt[3] = (__bf16)tanhf(v.w);
      *(v4bf*)&Ks[r * 64 + c4 * 4]  = pr;
      *(v4bf*)&Kts[r * 64 + c4 * 4] = pt;
    }
    #pragma unroll
    for (int i = 0; i < 32; i++) {
      int e  = tid + i * 128;
      int d  = e & 63, kv = e >> 6;
      Vts[d * 64 + kv] = (__bf16)Vbase[(long long)(j0 + kv) * E + d];
    }
    __syncthreads();

    // scores: S = Q K^T / 8 + lam * QJ tanh(K)^T
    v8f sacc[4], dacc[4];
    #pragma unroll
    for (int nf = 0; nf < 4; nf++) { sacc[nf] = vzero8(); dacc[nf] = vzero8(); }
    #pragma unroll
    for (int kk = 0; kk < 64; kk += 32) {
      v16bf aq = frag_ld(&Qs[(wv * 16) * 64 + kk], 64);
      v16bf aj = frag_ld(&QJs[(wv * 16) * 64 + kk], 64);
      #pragma unroll
      for (int nf = 0; nf < 4; nf++) {
        v16bf bk = frag_ld(&Ks[(nf * 16) * 64 + kk], 64);
        v16bf bt = frag_ld(&Kts[(nf * 16) * 64 + kk], 64);
        sacc[nf] = wmma_bf16(aq, bk, sacc[nf]);
        dacc[nf] = wmma_bf16(aj, bt, dacc[nf]);
      }
    }
    #pragma unroll
    for (int nf = 0; nf < 4; nf++) {
      #pragma unroll
      for (int i = 0; i < 8; i++) {
        int m = wv * 16 + ((lane >> 4) << 3) + i;
        int n = nf * 16 + (lane & 15);
        float sv = sacc[nf][i] * scl + lam * dacc[nf][i];
        if (j0 + n > q0 + m) sv = -3.0e38f;
        Ss[m * 64 + n] = sv;
      }
    }
    __syncthreads();

    // online softmax per query row
    if (tid < 64) {
      int r = tid;
      float mo = m_s[r];
      float mt = mo;
      for (int c = 0; c < 64; c++) mt = fmaxf(mt, Ss[r * 64 + c]);
      float scale = expf(mo - mt);
      float lsum  = l_s[r] * scale;
      for (int c = 0; c < 64; c++) {
        float pv = expf(Ss[r * 64 + c] - mt);
        lsum += pv;
        Ps[r * 64 + c] = (__bf16)pv;
      }
      m_s[r] = mt; l_s[r] = lsum; sc_s[r] = scale;
    }
    __syncthreads();

    // rescale running context, then acc += P @ V
    #pragma unroll
    for (int nf = 0; nf < 4; nf++) {
      #pragma unroll
      for (int i = 0; i < 8; i++) {
        int m = wv * 16 + ((lane >> 4) << 3) + i;
        acc[nf][i] *= sc_s[m];
      }
    }
    #pragma unroll
    for (int kk = 0; kk < 64; kk += 32) {
      v16bf ap = frag_ld(&Ps[(wv * 16) * 64 + kk], 64);
      #pragma unroll
      for (int nf = 0; nf < 4; nf++) {
        v16bf bvf = frag_ld(&Vts[(nf * 16) * 64 + kk], 64);
        acc[nf] = wmma_bf16(ap, bvf, acc[nf]);
      }
    }
    __syncthreads();
  }

  float* cb = ctx + (long long)(b * S + q0) * E + hh * Dh;
  #pragma unroll
  for (int nf = 0; nf < 4; nf++) {
    #pragma unroll
    for (int i = 0; i < 8; i++) {
      int m = wv * 16 + ((lane >> 4) << 3) + i;
      int d = nf * 16 + (lane & 15);
      cb[(long long)m * E + d] = acc[nf][i] / l_s[m];
    }
  }
}

// ---------------------------------------------------------------------------
// QBNN post: z = ht + lam * dj * tanh(ht); out = gelu(LN(z))
// ---------------------------------------------------------------------------
__global__ __launch_bounds__(256)
void qbnn_post_kernel(const float* __restrict__ ht, const float* __restrict__ dj,
                      const float* __restrict__ lb, const float* __restrict__ nw,
                      const float* __restrict__ nb, float* __restrict__ out, int N) {
  long long row = blockIdx.x;
  const float* hr = ht + row * N;
  const float* dr = dj + row * N;
  float* orow = out + row * N;
  float lbv = *lb;
  float sig = 1.f / (1.f + expf(-lbv));
  float lam = 0.25f + 0.5f * (sig * 0.7f + 0.15f);
  float s = 0.f, ss = 0.f;
  for (int c = threadIdx.x; c < N; c += 256) {
    float h = hr[c];
    float zv = h + lam * dr[c] * tanhf(h);
    s += zv; ss += zv * zv;
  }
  block_reduce2(s, ss);
  float mean = s / (float)N;
  float var  = ss / (float)N - mean * mean;
  float inv  = rsqrtf(var + 1e-5f);
  for (int c = threadIdx.x; c < N; c += 256) {
    float h = hr[c];
    float zv = h + lam * dr[c] * tanhf(h);
    float y = (zv - mean) * inv * nw[c] + nb[c];
    orow[c] = gelu_f(y);
  }
}

// ---------------------------------------------------------------------------
// Final combine: out = x1 + 0.7 * ffn_std + 0.3 * q2
// ---------------------------------------------------------------------------
__global__ __launch_bounds__(256)
void combine_kernel(const float* __restrict__ x1, const float* __restrict__ f,
                    const float* __restrict__ q, float* __restrict__ out, long long n) {
  long long i = (long long)blockIdx.x * 256 + threadIdx.x;
  if (i < n) out[i] = x1[i] + 0.7f * f[i] + 0.3f * q[i];
}

// ---------------------------------------------------------------------------
// Orchestration
// ---------------------------------------------------------------------------
extern "C" void kernel_launch(void* const* d_in, const int* in_sizes, int n_in,
                              void* d_out, int out_size, void* d_ws, size_t ws_size,
                              hipStream_t stream) {
  (void)in_sizes; (void)n_in; (void)out_size; (void)ws_size;
  const float* x    = (const float*)d_in[0];
  const float* n1w  = (const float*)d_in[1];
  const float* n1b  = (const float*)d_in[2];
  const float* wq   = (const float*)d_in[3];
  const float* bq   = (const float*)d_in[4];
  const float* wk   = (const float*)d_in[5];
  const float* bk   = (const float*)d_in[6];
  const float* wvw  = (const float*)d_in[7];
  const float* bv   = (const float*)d_in[8];
  const float* wo   = (const float*)d_in[9];
  const float* bo   = (const float*)d_in[10];
  const float* Jat  = (const float*)d_in[11];
  const float* lamA = (const float*)d_in[12];
  const float* n2w  = (const float*)d_in[13];
  const float* n2b  = (const float*)d_in[14];
  const float* wf1  = (const float*)d_in[15];
  const float* bf1  = (const float*)d_in[16];
  const float* wf2  = (const float*)d_in[17];
  const float* bf2  = (const float*)d_in[18];
  const float* wq1  = (const float*)d_in[19];
  const float* bq1  = (const float*)d_in[20];
  const float* J1   = (const float*)d_in[21];
  const float* lb1  = (const float*)d_in[22];
  const float* q1nw = (const float*)d_in[23];
  const float* q1nb = (const float*)d_in[24];
  const float* wq2  = (const float*)d_in[25];
  const float* bq2  = (const float*)d_in[26];
  const float* J2   = (const float*)d_in[27];
  const float* lb2  = (const float*)d_in[28];
  const float* q2nw = (const float*)d_in[29];
  const float* q2nb = (const float*)d_in[30];
  float* out = (float*)d_out;

  const long long M4  = 4LL * 1024 * 1024;    // B*S*E
  const long long M16 = 16LL * 1024 * 1024;   // B*S*FF
  float* p    = (float*)d_ws;
  float* h    = p; p += M4;
  float* Qb   = p; p += M4;
  float* Kb   = p; p += M4;
  float* Vb   = p; p += M4;
  float* QJ   = p; p += M4;
  float* ctx  = p; p += M4;
  float* x1   = p; p += M4;
  float* h2   = p; p += M4;
  float* ffs  = p; p += M4;
  float* ht2  = p; p += M4;
  float* dj2  = p; p += M4;
  float* q2   = p; p += M4;
  float* big1 = p; p += M16;   // t_ff1, later q1
  float* big2 = p; p += M16;   // ht1
  float* big3 = p; p += M16;   // dj1

  const int Mrows = 4096;      // B*S
  dim3 blk(256);
  dim3 gEE(1024 / 128, Mrows / 128, 1);
  dim3 gEF(4096 / 128, Mrows / 128, 1);

  // 1. LN1
  ln_kernel<<<Mrows, blk, 0, stream>>>(x, n1w, n1b, h, 1024);
  // 2. Q,K,V projections
  gemm_wmma_kernel<0,0><<<gEE, blk, 0, stream>>>(h, wq,  bq, nullptr, Qb, Mrows,1024,1024, 1024,1024,1024, 1,0,0,0,0);
  gemm_wmma_kernel<0,0><<<gEE, blk, 0, stream>>>(h, wk,  bk, nullptr, Kb, Mrows,1024,1024, 1024,1024,1024, 1,0,0,0,0);
  gemm_wmma_kernel<0,0><<<gEE, blk, 0, stream>>>(h, wvw, bv, nullptr, Vb, Mrows,1024,1024, 1024,1024,1024, 1,0,0,0,0);
  // 3. QJ[b,h] = tanh(Q[b,h]) @ J_attn[h]   (batched over z = b*16+h)
  {
    dim3 gQJ(1, 1024 / 128, 64);
    gemm_wmma_kernel<1,0><<<gQJ, blk, 0, stream>>>(Qb, Jat, nullptr, nullptr, QJ,
        1024, 64, 64, 1024, 64, 64,
        16, (long long)1024 * 1024, 64, 4096, (long long)1024 * 64);
  }
  // 4. attention (flash-style, causal, QBNN delta)
  {
    dim3 gAT(16, 64);
    attn_kernel<<<gAT, dim3(128), 0, stream>>>(Qb, QJ, Kb, Vb, lamA, ctx);
  }
  // 5. out projection + residual: x1 = x + ctx @ wo + bo
  gemm_wmma_kernel<0,2><<<gEE, blk, 0, stream>>>(ctx, wo, bo, x, x1, Mrows,1024,1024, 1024,1024,1024, 1,0,0,0,0);
  // 6. LN2
  ln_kernel<<<Mrows, blk, 0, stream>>>(x1, n2w, n2b, h2, 1024);
  // 7. FFN std: big1 = gelu(h2 @ wf1 + bf1)
  gemm_wmma_kernel<0,1><<<gEF, blk, 0, stream>>>(h2, wf1, bf1, nullptr, big1, Mrows,4096,1024, 1024,4096,4096, 1,0,0,0,0);
  // 8. ffs = big1 @ wf2 + bf2
  gemm_wmma_kernel<0,0><<<gEE, blk, 0, stream>>>(big1, wf2, bf2, nullptr, ffs, Mrows,1024,4096, 4096,1024,1024, 1,0,0,0,0);
  // 9. QBNN layer 1 (input h2, width FF)
  gemm_wmma_kernel<0,0><<<gEF, blk, 0, stream>>>(h2, wq1, bq1, nullptr, big2, Mrows,4096,1024, 1024,4096,4096, 1,0,0,0,0);
  gemm_wmma_kernel<1,0><<<gEF, blk, 0, stream>>>(h2, J1, nullptr, nullptr, big3, Mrows,4096,1024, 1024,4096,4096, 1,0,0,0,0);
  qbnn_post_kernel<<<Mrows, blk, 0, stream>>>(big2, big3, lb1, q1nw, q1nb, big1, 4096);
  // 10. QBNN layer 2 (input q1=big1, width E)
  gemm_wmma_kernel<0,0><<<gEE, blk, 0, stream>>>(big1, wq2, bq2, nullptr, ht2, Mrows,1024,4096, 4096,1024,1024, 1,0,0,0,0);
  gemm_wmma_kernel<1,0><<<gEE, blk, 0, stream>>>(big1, J2, nullptr, nullptr, dj2, Mrows,1024,4096, 4096,1024,1024, 1,0,0,0,0);
  qbnn_post_kernel<<<Mrows, blk, 0, stream>>>(ht2, dj2, lb2, q2nw, q2nb, q2, 1024);
  // 11. combine
  combine_kernel<<<(int)((M4 + 255) / 256), blk, 0, stream>>>(x1, ffs, q2, out, M4);
}
